// RIMCell_65635690218140
// MI455X (gfx1250) — matrix operand order repaired and checked by reading, not compiled, not run on hardware
//
#include <hip/hip_runtime.h>
#include <hip/hip_bf16.h>
#include <math.h>

// ---------------------------------------------------------------------------
// RIM cell forward for MI455X (gfx1250, wave32, WMMA 16x16x32 bf16)
// B=4096, D=512, H=512, U=8, K=5, IV=400, NH=4, CK=32, CV=100
//
// ~132 GFLOP total, dominated by batch-tiled GEMMs (M=16 rows/tile).
// Weights (~60MB) stay L2-resident (192MB L2); per-call prep pass converts
// them once to bf16 in WMMA-friendly [N][Kpad32] layout (zero-padded K) so the
// hot loops contain no bounds checks and stream B operands as b128 loads.
// ---------------------------------------------------------------------------

typedef __attribute__((ext_vector_type(16))) __bf16 v16bf;
typedef __attribute__((ext_vector_type(8)))  __bf16 v8bf;
typedef __attribute__((ext_vector_type(8)))  float  v8f;

__device__ __forceinline__ v8f zero8() {
    v8f z;
#pragma unroll
    for (int j = 0; j < 8; ++j) z[j] = 0.0f;
    return z;
}

__device__ __forceinline__ float sigmoidf_(float x) { return 1.0f / (1.0f + __expf(-x)); }

// D = A(16x32) * B(32x16) + C, bf16 in / f32 out
__device__ __forceinline__ v8f wmma_bf16(v16bf a, v16bf b, v8f c) {
    return __builtin_amdgcn_wmma_f32_16x16x32_bf16(false, a, false, b, (short)0, c, false, false);
}

// A fragment (16x32 bf16) from LDS row-major [16][lda].
// ISA 16-bit A layout: lane L: m=L%16, g=L/16; element i -> K = (i&7)+8g+16*(i>>3)
// => per lane two contiguous 8xbf16 (16B) chunks at row+{0,16} -> 2x ds_load_b128.
__device__ __forceinline__ v16bf ldsA_bf16(const __bf16* A, int lda, int k0, int lane) {
    int m = lane & 15, g = lane >> 4;
    const __bf16* row = A + m * lda + k0 + 8 * g;
    v8bf lo = *(const v8bf*)(row);
    v8bf hi = *(const v8bf*)(row + 16);
    v16bf a;
#pragma unroll
    for (int i = 0; i < 8; ++i) { a[i] = lo[i]; a[i + 8] = hi[i]; }
    return a;
}

// B fragment (32x16) from global bf16 W stored [N][ldK] (K contiguous, padded).
// B layout: lane L: n=L%16, g=L/16; element i -> K = (i&7)+8g+16*(i>>3)
// => per lane two 8xbf16 (16B) chunks at +{0,16} -> 2x global_load_b128.
__device__ __forceinline__ v16bf gB_bf16(const __bf16* W, int ldK, int n0, int k0, int lane) {
    int n  = n0 + (lane & 15);
    int kb = k0 + 8 * (lane >> 4);
    const v8bf* p = (const v8bf*)(W + (size_t)n * ldK + kb);
    v8bf lo = p[0], hi = p[2];
    v16bf b;
#pragma unroll
    for (int i = 0; i < 8; ++i) { b[i] = lo[i]; b[i + 8] = hi[i]; }
    return b;
}

// ---------------------------------------------------------------------------
// Weight prep: f32 -> bf16, into [U][N][ldK] with K zero-padded to ldK.
// Clamp-address + value-select so no per-element exec-mask branches.
// ---------------------------------------------------------------------------
__global__ __launch_bounds__(256) void w_conv_nk(   // src already [U][N][Ksrc]
    const float* __restrict__ src, __bf16* __restrict__ dst, int N, int Ksrc, int ldK)
{
    int u = blockIdx.y;
    const float* s = src + (size_t)u * N * Ksrc;
    __bf16* d = dst + (size_t)u * N * ldK;
    size_t total = (size_t)N * ldK;
    for (size_t i = (size_t)blockIdx.x * 256 + threadIdx.x; i < total; i += (size_t)gridDim.x * 256) {
        int n = (int)(i / ldK), k = (int)(i % ldK);
        int kc = k < Ksrc ? k : (Ksrc - 1);
        float v = s[(size_t)n * Ksrc + kc];
        d[i] = (k < Ksrc) ? (__bf16)v : (__bf16)0.0f;
    }
}

__global__ __launch_bounds__(256) void w_conv_kn(   // src [U][Ksrc][N] -> transpose
    const float* __restrict__ src, __bf16* __restrict__ dst, int Ksrc, int N, int ldK)
{
    int u = blockIdx.y;
    const float* s = src + (size_t)u * Ksrc * N;
    __bf16* d = dst + (size_t)u * N * ldK;
    size_t total = (size_t)N * ldK;
    for (size_t i = (size_t)blockIdx.x * 256 + threadIdx.x; i < total; i += (size_t)gridDim.x * 256) {
        int n = (int)(i / ldK), k = (int)(i % ldK);
        int kc = k < Ksrc ? k : (Ksrc - 1);
        float v = s[(size_t)kc * N + n];
        d[i] = (k < Ksrc) ? (__bf16)v : (__bf16)0.0f;
    }
}

// ---------------------------------------------------------------------------
// Kernel 1: input attention gating.  One block per batch row (VALU; ~4 GFLOP).
// ---------------------------------------------------------------------------
__global__ __launch_bounds__(256) void k1_gate(
    const float* __restrict__ x, const float* __restrict__ hs,
    const float* __restrict__ key_w, const float* __restrict__ key_b,
    const float* __restrict__ value_w, const float* __restrict__ value_b,
    const float* __restrict__ query_w,
    float* __restrict__ ws_mask, float* __restrict__ ws_inp)
{
    const int b = blockIdx.x;
    const int tid = threadIdx.x;
    __shared__ float xs[512];
    __shared__ float hss[8 * 512];
    __shared__ float k0[64];
    __shared__ float v0[400];
    __shared__ float ql[8 * 64];
    __shared__ float s0[8], s1[8], p0s[8], p1s[8], msk[8];

    for (int i = tid; i < 512; i += 256)  xs[i]  = x[(size_t)b * 512 + i];
    for (int i = tid; i < 4096; i += 256) hss[i] = hs[(size_t)b * 4096 + i];
    __syncthreads();

    for (int c = tid; c < 64; c += 256) {
        float acc = key_b[c];
        for (int d = 0; d < 512; ++d) acc += xs[d] * key_w[d * 64 + c];
        k0[c] = acc;
    }
    for (int c = tid; c < 400; c += 256) {
        float acc = value_b[c];
        for (int d = 0; d < 512; ++d) acc += xs[d] * value_w[d * 400 + c];
        v0[c] = acc;
    }
    for (int i = tid; i < 512; i += 256) {
        int u = i >> 6, c = i & 63;
        const float* qw = query_w + (size_t)u * 512 * 64;
        float acc = 0.f;
        for (int h = 0; h < 512; ++h) acc += hss[u * 512 + h] * qw[h * 64 + c];
        ql[i] = acc;
    }
    __syncthreads();
    if (tid < 8) {
        float a0 = 0.f, a1 = 0.f;
        for (int c = 0; c < 64; ++c) { a0 += ql[tid * 64 + c] * k0[c]; a1 += ql[tid * 64 + c] * key_b[c]; }
        s0[tid] = a0 * 0.125f;  // 1/sqrt(64)
        s1[tid] = a1 * 0.125f;
    }
    __syncthreads();
    if (tid < 8) {
        int rank = 0;   // top-5 of 8, ties to lower index (matches top_k)
        for (int v = 0; v < 8; ++v)
            if (s0[v] > s0[tid] || (s0[v] == s0[tid] && v < tid)) rank++;
        msk[tid] = (rank < 5) ? 1.0f : 0.0f;
        float mx = fmaxf(s0[tid], s1[tid]);
        float e0 = __expf(s0[tid] - mx), e1 = __expf(s1[tid] - mx);
        float inv = 1.0f / (e0 + e1);
        p0s[tid] = e0 * inv;
        p1s[tid] = e1 * inv;
        ws_mask[b * 8 + tid] = msk[tid];
    }
    __syncthreads();
    for (int i = tid; i < 8 * 400; i += 256) {
        int u = i / 400, v = i - u * 400;
        ws_inp[(size_t)b * 3200 + i] = (p0s[u] * v0[v] + p1s[u] * value_b[v]) * msk[u];
    }
}

// ---------------------------------------------------------------------------
// Kernel 2: GRU.  Block = (unit u, 16-row tile), 8 waves.
// gi = inp @ Wi_b^T (K=416), gh = hm @ Wh_b^T (K=512); 6 accumulators/wave so
// the GRU gates fuse in-register on the WMMA outputs.  Then LN -> x_final.
// ---------------------------------------------------------------------------
__global__ __launch_bounds__(256) void k2_gru(
    const float* __restrict__ hs, const float* __restrict__ h_masks,
    const __bf16* __restrict__ Wi_all, const float* __restrict__ b_ih,
    const __bf16* __restrict__ Wh_all, const float* __restrict__ b_hh,
    const float* __restrict__ rnn_ln_g, const float* __restrict__ rnn_ln_b,
    const float* __restrict__ ws_inp, float* __restrict__ ws_hnew,
    float* __restrict__ out_xfinal)
{
    const int u = blockIdx.x;
    const int tile = blockIdx.y;
    const int tid = threadIdx.x;
    const int lane = tid & 31, wave = tid >> 5;
    const int r0 = tile * 16;

    __shared__ __bf16 sInp[16 * 416];   // inp tile, K padded 400->416 (zeros)
    __shared__ __bf16 sHmb[16 * 512];   // hm tile, bf16 (A operand for gh)
    __shared__ float  sF[16 * 512];     // hm f32, then reused for h_new
    __shared__ float  sMean[16], sInv[16];

    for (int i = tid; i < 16 * 400; i += 256) {
        int m = i / 400, c = i - m * 400;
        sInp[m * 416 + c] = (__bf16)ws_inp[((size_t)(r0 + m) * 8 + u) * 400 + c];
    }
    for (int i = tid; i < 16 * 16; i += 256) {
        int m = i >> 4, c = 400 + (i & 15);
        sInp[m * 416 + c] = (__bf16)0.0f;
    }
    for (int i = tid; i < 16 * 512; i += 256) {
        int m = i >> 9, h = i & 511;
        float v = hs[((size_t)(r0 + m) * 8 + u) * 512 + h] * h_masks[r0 + m];
        sF[i] = v;
        sHmb[i] = (__bf16)v;
    }
    __syncthreads();

    const __bf16* Wi = Wi_all + (size_t)u * 1536 * 416;
    const __bf16* Wh = Wh_all + (size_t)u * 1536 * 512;
    const float*  bi = b_ih + u * 1536;
    const float*  bh = b_hh + u * 1536;

    v8f hnewReg[4];
#pragma unroll
    for (int t = 0; t < 4; ++t) {
        const int n0 = wave * 64 + t * 16;
        v8f ir = zero8(), iz = zero8(), in_ = zero8();
        v8f hr = zero8(), hz = zero8(), hn = zero8();
        for (int ks = 0; ks < 13; ++ks) {                 // gi, K = 416
            v16bf a  = ldsA_bf16(sInp, 416, ks * 32, lane);
            v16bf br = gB_bf16(Wi, 416, n0,        ks * 32, lane);
            v16bf bz = gB_bf16(Wi, 416, n0 + 512,  ks * 32, lane);
            v16bf bn = gB_bf16(Wi, 416, n0 + 1024, ks * 32, lane);
            ir  = wmma_bf16(a, br, ir);
            iz  = wmma_bf16(a, bz, iz);
            in_ = wmma_bf16(a, bn, in_);
        }
        for (int ks = 0; ks < 16; ++ks) {                 // gh, K = 512
            v16bf a  = ldsA_bf16(sHmb, 512, ks * 32, lane);
            v16bf br = gB_bf16(Wh, 512, n0,        ks * 32, lane);
            v16bf bz = gB_bf16(Wh, 512, n0 + 512,  ks * 32, lane);
            v16bf bn = gB_bf16(Wh, 512, n0 + 1024, ks * 32, lane);
            hr = wmma_bf16(a, br, hr);
            hz = wmma_bf16(a, bz, hz);
            hn = wmma_bf16(a, bn, hn);
        }
        const int n  = n0 + (lane & 15);
        const int mb = 8 * (lane >> 4);
        const float bir = bi[n], biz = bi[n + 512], bin_ = bi[n + 1024];
        const float bhr = bh[n], bhz = bh[n + 512], bhn  = bh[n + 1024];
        v8f hv;
#pragma unroll
        for (int j = 0; j < 8; ++j) {
            float r  = sigmoidf_(ir[j] + bir + hr[j] + bhr);
            float z  = sigmoidf_(iz[j] + biz + hz[j] + bhz);
            float nn = tanhf(in_[j] + bin_ + r * (hn[j] + bhn));
            float hm = sF[(mb + j) * 512 + n];
            hv[j] = (1.0f - z) * nn + z * hm;
        }
        hnewReg[t] = hv;
    }
    __syncthreads();   // all waves done reading hm
#pragma unroll
    for (int t = 0; t < 4; ++t) {
        const int n  = wave * 64 + t * 16 + (lane & 15);
        const int mb = 8 * (lane >> 4);
#pragma unroll
        for (int j = 0; j < 8; ++j) sF[(mb + j) * 512 + n] = hnewReg[t][j];
    }
    __syncthreads();
    for (int i = tid; i < 16 * 512; i += 256) {
        int m = i >> 9, h = i & 511;
        ws_hnew[((size_t)(r0 + m) * 8 + u) * 512 + h] = sF[i];
    }
    if (tid < 16) {     // row layernorm stats
        float s = 0.f;
        for (int h = 0; h < 512; ++h) s += sF[tid * 512 + h];
        float mean = s * (1.0f / 512.0f);
        float vv = 0.f;
        for (int h = 0; h < 512; ++h) { float d = sF[tid * 512 + h] - mean; vv += d * d; }
        sMean[tid] = mean;
        sInv[tid]  = rsqrtf(vv * (1.0f / 512.0f) + 1e-5f);
    }
    __syncthreads();
    for (int i = tid; i < 16 * 512; i += 256) {
        int m = i >> 9, h = i & 511;
        float y = (sF[i] - sMean[m]) * sInv[m] * rnn_ln_g[u * 512 + h] + rnn_ln_b[u * 512 + h];
        out_xfinal[(size_t)(r0 + m) * 4096 + u * 512 + h] = y;
    }
}

// ---------------------------------------------------------------------------
// Kernel 3: qc/kc/vc projections.  Block = (unit u, 16-row tile), 8 waves.
// 41 N-tiles (8 qc + 8 kc + 25 vc), K = 512, weights pre-transposed to [N][512].
// ---------------------------------------------------------------------------
__global__ __launch_bounds__(256) void k3_qkv(
    const float* __restrict__ ws_hnew,
    const __bf16* __restrict__ Wq_all, const __bf16* __restrict__ Wk_all,
    const __bf16* __restrict__ Wv_all,
    float* __restrict__ ws_qc, float* __restrict__ ws_kc, float* __restrict__ ws_vc)
{
    const int u = blockIdx.x, tile = blockIdx.y;
    const int tid = threadIdx.x, lane = tid & 31, wave = tid >> 5;
    const int r0 = tile * 16;
    __shared__ __bf16 sA[16 * 512];
    for (int i = tid; i < 16 * 512; i += 256) {
        int m = i >> 9, h = i & 511;
        sA[i] = (__bf16)ws_hnew[((size_t)(r0 + m) * 8 + u) * 512 + h];
    }
    __syncthreads();
    const __bf16* Wq = Wq_all + (size_t)u * 128 * 512;
    const __bf16* Wk = Wk_all + (size_t)u * 128 * 512;
    const __bf16* Wv = Wv_all + (size_t)u * 400 * 512;
    for (int tix = wave; tix < 41; tix += 8) {
        const __bf16* W; float* dst; int ldN, n0;
        if (tix < 8)       { W = Wq; dst = ws_qc; ldN = 128; n0 = tix * 16; }
        else if (tix < 16) { W = Wk; dst = ws_kc; ldN = 128; n0 = (tix - 8) * 16; }
        else               { W = Wv; dst = ws_vc; ldN = 400; n0 = (tix - 16) * 16; }
        v8f acc = zero8();
        for (int ks = 0; ks < 16; ++ks) {
            v16bf a = ldsA_bf16(sA, 512, ks * 32, lane);
            v16bf b = gB_bf16(W, 512, n0, ks * 32, lane);
            acc = wmma_bf16(a, b, acc);
        }
        const int n  = n0 + (lane & 15);
        const int mb = 8 * (lane >> 4);
#pragma unroll
        for (int j = 0; j < 8; ++j)
            dst[((size_t)(r0 + mb + j) * 8 + u) * ldN + n] = acc[j];
    }
}

// ---------------------------------------------------------------------------
// Kernel 4: communication attention (4 heads, 8x8 per row — tiny, VALU).
// ---------------------------------------------------------------------------
__global__ __launch_bounds__(256) void k4_att(
    const float* __restrict__ ws_qc, const float* __restrict__ ws_kc,
    const float* __restrict__ ws_vc, const float* __restrict__ ws_mask,
    float* __restrict__ ws_ctx)
{
    const int b = blockIdx.x, tid = threadIdx.x;
    __shared__ float q[8 * 128], kk[8 * 128], vv[8 * 400], ap[256], msk[8];
    for (int i = tid; i < 8 * 128; i += 256) {
        q[i]  = ws_qc[(size_t)b * 1024 + i];
        kk[i] = ws_kc[(size_t)b * 1024 + i];
    }
    for (int i = tid; i < 8 * 400; i += 256) vv[i] = ws_vc[(size_t)b * 3200 + i];
    if (tid < 8) msk[tid] = ws_mask[b * 8 + tid];
    __syncthreads();
    {
        int n = tid >> 6, uq = (tid >> 3) & 7, v = tid & 7;
        float acc = 0.f;
        for (int c = 0; c < 32; ++c) acc += q[uq * 128 + n * 32 + c] * kk[v * 128 + n * 32 + c];
        ap[tid] = acc * 0.17677669529663687f;   // 1/sqrt(32)
    }
    __syncthreads();
    if (tid < 32) {
        int n = tid >> 3, uq = tid & 7;
        float mx = -1e30f;
        for (int v = 0; v < 8; ++v) mx = fmaxf(mx, ap[n * 64 + uq * 8 + v]);
        float e[8], s = 0.f;
        for (int v = 0; v < 8; ++v) { e[v] = __expf(ap[n * 64 + uq * 8 + v] - mx); s += e[v]; }
        float inv = msk[uq] / s;
        for (int v = 0; v < 8; ++v) ap[n * 64 + uq * 8 + v] = e[v] * inv;
    }
    __syncthreads();
    for (int i = tid; i < 8 * 400; i += 256) {
        int uq = i / 400, col = i - uq * 400;
        int n = col / 100;
        float acc = 0.f;
        for (int v = 0; v < 8; ++v) acc += ap[n * 64 + uq * 8 + v] * vv[v * 400 + col];
        ws_ctx[(size_t)b * 3200 + i] = acc;
    }
}

// ---------------------------------------------------------------------------
// Kernel 5: output projection + residual + LN + mask-select.
// ctx(16x416 padded) @ Wo_b[u]([512][416] padded) -> (16x512).
// ---------------------------------------------------------------------------
__global__ __launch_bounds__(256) void k5_out(
    const float* __restrict__ ws_ctx, const float* __restrict__ ws_hnew,
    const float* __restrict__ ws_mask, const float* __restrict__ hs,
    const __bf16* __restrict__ Wo_all, const float* __restrict__ ln_g,
    const float* __restrict__ ln_b, float* __restrict__ out_hs)
{
    const int u = blockIdx.x, tile = blockIdx.y;
    const int tid = threadIdx.x, lane = tid & 31, wave = tid >> 5;
    const int r0 = tile * 16;
    __shared__ __bf16 sA[16 * 416];
    __shared__ float  sC[16 * 512];
    __shared__ float  sMean[16], sInv[16];
    for (int i = tid; i < 16 * 400; i += 256) {
        int m = i / 400, c = i - m * 400;
        sA[m * 416 + c] = (__bf16)ws_ctx[((size_t)(r0 + m) * 8 + u) * 400 + c];
    }
    for (int i = tid; i < 16 * 16; i += 256) {
        int m = i >> 4, c = 400 + (i & 15);
        sA[m * 416 + c] = (__bf16)0.0f;
    }
    __syncthreads();
    const __bf16* Wo = Wo_all + (size_t)u * 512 * 416;
#pragma unroll
    for (int t = 0; t < 4; ++t) {
        const int n0 = wave * 64 + t * 16;
        v8f acc = zero8();
        for (int ks = 0; ks < 13; ++ks) {
            v16bf a = ldsA_bf16(sA, 416, ks * 32, lane);
            v16bf b = gB_bf16(Wo, 416, n0, ks * 32, lane);
            acc = wmma_bf16(a, b, acc);
        }
        const int n  = n0 + (lane & 15);
        const int mb = 8 * (lane >> 4);
#pragma unroll
        for (int j = 0; j < 8; ++j) {
            float hb = ws_hnew[((size_t)(r0 + mb + j) * 8 + u) * 512 + n];
            sC[(mb + j) * 512 + n] = acc[j] + hb;
        }
    }
    __syncthreads();
    if (tid < 16) {
        float s = 0.f;
        for (int h = 0; h < 512; ++h) s += sC[tid * 512 + h];
        float mean = s * (1.0f / 512.0f);
        float vv = 0.f;
        for (int h = 0; h < 512; ++h) { float d = sC[tid * 512 + h] - mean; vv += d * d; }
        sMean[tid] = mean;
        sInv[tid]  = rsqrtf(vv * (1.0f / 512.0f) + 1e-5f);
    }
    __syncthreads();
    for (int i = tid; i < 16 * 512; i += 256) {
        int m = i >> 9, h = i & 511;
        float hcomm = (sC[i] - sMean[m]) * sInv[m] * ln_g[h] + ln_b[h];
        float mk   = ws_mask[(r0 + m) * 8 + u];
        float hold = hs[((size_t)(r0 + m) * 8 + u) * 512 + h];
        out_hs[((size_t)(r0 + m) * 8 + u) * 512 + h] = mk * hcomm + (1.0f - mk) * hold;
    }
}

// ---------------------------------------------------------------------------
extern "C" void kernel_launch(void* const* d_in, const int* in_sizes, int n_in,
                              void* d_out, int out_size, void* d_ws, size_t ws_size,
                              hipStream_t stream)
{
    const float* x        = (const float*)d_in[0];
    const float* hs       = (const float*)d_in[1];
    const float* h_masks  = (const float*)d_in[2];
    const float* key_w    = (const float*)d_in[3];
    const float* key_b    = (const float*)d_in[4];
    const float* value_w  = (const float*)d_in[5];
    const float* value_b  = (const float*)d_in[6];
    const float* query_w  = (const float*)d_in[7];
    const float* qc_w     = (const float*)d_in[8];
    const float* kc_w     = (const float*)d_in[9];
    const float* vc_w     = (const float*)d_in[10];
    const float* out_w    = (const float*)d_in[11];
    const float* ln_g     = (const float*)d_in[12];
    const float* ln_b     = (const float*)d_in[13];
    const float* W_ih     = (const float*)d_in[14];
    const float* b_ih     = (const float*)d_in[15];
    const float* W_hh     = (const float*)d_in[16];
    const float* b_hh     = (const float*)d_in[17];
    const float* rnn_ln_g = (const float*)d_in[18];
    const float* rnn_ln_b = (const float*)d_in[19];
    (void)in_sizes; (void)n_in; (void)out_size; (void)ws_size;

    float* out_xfinal = (float*)d_out;                      // (B, U*H)
    float* out_hs     = out_xfinal + (size_t)4096 * 4096;   // (B, U, H)

    char* wsb = (char*)d_ws;
    size_t off = 0;
    auto walloc = [&](size_t bytes) -> void* {
        void* p = (void*)(wsb + off);
        off = (off + bytes + 255) & ~(size_t)255;
        return p;
    };
    float*  ws_mask = (float*)walloc((size_t)4096 * 8 * 4);
    float*  ws_inp  = (float*)walloc((size_t)4096 * 8 * 400 * 4);   // reused as ctx
    float*  ws_hnew = (float*)walloc((size_t)4096 * 8 * 512 * 4);
    float*  ws_qc   = (float*)walloc((size_t)4096 * 8 * 128 * 4);
    float*  ws_kc   = (float*)walloc((size_t)4096 * 8 * 128 * 4);
    float*  ws_vc   = (float*)walloc((size_t)4096 * 8 * 400 * 4);
    __bf16* ws_wi   = (__bf16*)walloc((size_t)8 * 1536 * 416 * 2);  // [u][1536][416]
    __bf16* ws_wh   = (__bf16*)walloc((size_t)8 * 1536 * 512 * 2);  // [u][1536][512]
    __bf16* ws_wq   = (__bf16*)walloc((size_t)8 * 128 * 512 * 2);   // [u][128][512]
    __bf16* ws_wk   = (__bf16*)walloc((size_t)8 * 128 * 512 * 2);   // [u][128][512]
    __bf16* ws_wv   = (__bf16*)walloc((size_t)8 * 400 * 512 * 2);   // [u][400][512]
    __bf16* ws_wo   = (__bf16*)walloc((size_t)8 * 512 * 416 * 2);   // [u][512][416]

    // --- weight prep (bf16, WMMA-B layout, zero-padded K) ---
    w_conv_nk<<<dim3(256, 8), dim3(256), 0, stream>>>(W_ih, ws_wi, 1536, 400, 416);
    w_conv_nk<<<dim3(256, 8), dim3(256), 0, stream>>>(W_hh, ws_wh, 1536, 512, 512);
    w_conv_kn<<<dim3(64, 8),  dim3(256), 0, stream>>>(qc_w, ws_wq, 512, 128, 512);
    w_conv_kn<<<dim3(64, 8),  dim3(256), 0, stream>>>(kc_w, ws_wk, 512, 128, 512);
    w_conv_kn<<<dim3(128, 8), dim3(256), 0, stream>>>(vc_w, ws_wv, 512, 400, 512);
    w_conv_kn<<<dim3(128, 8), dim3(256), 0, stream>>>(out_w, ws_wo, 400, 512, 416);

    // --- pipeline ---
    k1_gate<<<dim3(4096), dim3(256), 0, stream>>>(x, hs, key_w, key_b, value_w, value_b,
                                                  query_w, ws_mask, ws_inp);
    k2_gru<<<dim3(8, 256), dim3(256), 0, stream>>>(hs, h_masks, ws_wi, b_ih, ws_wh, b_hh,
                                                   rnn_ln_g, rnn_ln_b, ws_inp, ws_hnew, out_xfinal);
    k3_qkv<<<dim3(8, 256), dim3(256), 0, stream>>>(ws_hnew, ws_wq, ws_wk, ws_wv,
                                                   ws_qc, ws_kc, ws_vc);
    k4_att<<<dim3(4096), dim3(256), 0, stream>>>(ws_qc, ws_kc, ws_vc, ws_mask, ws_inp /*ctx*/);
    k5_out<<<dim3(8, 256), dim3(256), 0, stream>>>(ws_inp /*ctx*/, ws_hnew, ws_mask, hs,
                                                   ws_wo, ln_g, ln_b, out_hs);
}
